// causal_attention_2808908612117
// MI455X (gfx1250) — matrix operand order repaired and checked by reading, not compiled, not run on hardware
//
#include <hip/hip_runtime.h>
#include <hip/hip_bf16.h>

typedef unsigned short ushort_t;
typedef __attribute__((ext_vector_type(16))) __bf16   v16bf;
typedef __attribute__((ext_vector_type(8)))  float    v8f;
typedef __attribute__((ext_vector_type(4)))  unsigned int u32x4;
typedef __attribute__((ext_vector_type(8)))  int      i32x8;
typedef __attribute__((ext_vector_type(4)))  int      i32x4;

#define B_   2
#define T_   4096
#define E_   768
#define NH_  12
#define HS_  64
#define M_   (B_ * T_)      // 8192 rows
#define E3_  (3 * E_)       // 2304
#define QKV_SLOT ((size_t)B_ * NH_ * T_ * HS_)   // elements per q/k/v slot

__device__ __forceinline__ ushort_t f2bf(float f) {
  unsigned u = __builtin_bit_cast(unsigned, f);
  unsigned r = u + 0x7FFFu + ((u >> 16) & 1u);   // round-to-nearest-even
  return (ushort_t)(r >> 16);
}

__device__ __forceinline__ v16bf mkfrag(u32x4 lo, u32x4 hi) {
  union { u32x4 u[2]; v16bf v; } c;
  c.u[0] = lo; c.u[1] = hi;
  return c.v;
}

__device__ __forceinline__ unsigned lds_off(const void* p) {
  // LDS aperture: low 32 bits of the flat address are the LDS offset
  return (unsigned)(unsigned long long)(uintptr_t)p;
}

// ---- Tensor Data Mover: 2D bf16 tile load, D# per ISA cdna5 ch.8 ----
// tile_d0 elems contiguous (dim0), tile_d1 rows, row stride = stride0 elems.
__device__ __forceinline__ void tdm_load_2d(unsigned lds_addr, const void* gptr,
                                            unsigned tensor_d0, unsigned tensor_d1,
                                            unsigned tile_d0, unsigned tile_d1,
                                            unsigned stride0) {
  unsigned long long ga = (unsigned long long)(uintptr_t)gptr;
  u32x4 g0;
  g0[0] = 1u;                                       // count=1, user descriptor
  g0[1] = lds_addr;                                 // lds_addr [63:32]
  g0[2] = (unsigned)(ga & 0xFFFFFFFFu);             // global_addr [95:64]
  g0[3] = (unsigned)((ga >> 32) & 0x01FFFFFFu)      // global_addr [120:96]
          | (2u << 30);                             // type=2 ("image") [127:126]
  i32x8 g1;
  g1[0] = (int)(1u << 16);                          // wg_mask=0, data_size=1 (2B)
  g1[1] = (int)((tensor_d0 & 0xFFFFu) << 16);       // tensor_dim0[15:0] @ [63:48]
  g1[2] = (int)((tensor_d0 >> 16) | ((tensor_d1 & 0xFFFFu) << 16));
  g1[3] = (int)((tensor_d1 >> 16) | (tile_d0 << 16));   // tile_dim0 @ [127:112]
  g1[4] = (int)tile_d1;                             // tile_dim1; tile_dim2=0
  g1[5] = (int)stride0;                             // tensor_dim0_stride[31:0]
  g1[6] = 0;                                        // stride0 hi, stride1 lo
  g1[7] = 0;
  i32x4 z4 = {0, 0, 0, 0};
  i32x8 z8 = {0, 0, 0, 0, 0, 0, 0, 0};
  // 6-arg form (clang-23 / therock-10.0 lane)
  __builtin_amdgcn_tensor_load_to_lds(g0, g1, z4, z4, z8, 0);
}

// ---------------- conversion kernels ----------------

__global__ void k_f32_to_bf16(const float* __restrict__ src,
                              ushort_t* __restrict__ dst, int n) {
  for (int i = blockIdx.x * blockDim.x + threadIdx.x; i < n;
       i += gridDim.x * blockDim.x)
    dst[i] = f2bf(src[i]);
}

// src[rows][cols] (f32) -> dst[cols][rows] (bf16)
__global__ void k_transpose_bf16(const float* __restrict__ src,
                                 ushort_t* __restrict__ dst,
                                 int rows, int cols) {
  int n = rows * cols;
  for (int i = blockIdx.x * blockDim.x + threadIdx.x; i < n;
       i += gridDim.x * blockDim.x) {
    int r = i / cols, c = i - r * cols;
    dst[(size_t)c * rows + r] = f2bf(src[i]);
  }
}

// ---------------- QKV projection GEMM ----------------
// C[M_ x E3_] = Xb[M_ x E_] * WaT^T  (WaT stored [E3_][E_] row-major)
// Block tile 128x64, 8 waves in 4x2 grid, 32x32 per wave (4 WMMA / k-step).
// LDS tiles staged by the Tensor Data Mover, double-buffered.
__global__ __launch_bounds__(256) void k_qkv_gemm(
    const ushort_t* __restrict__ xb, const ushort_t* __restrict__ WaT,
    const float* __restrict__ b_attn, ushort_t* __restrict__ qkv) {
  __shared__ __align__(16) u32x4 At[2][128 * 4];   // 128 rows x 32 k (bf16)
  __shared__ __align__(16) u32x4 Bt[2][64 * 4];    // 64 rows x 32 k

  const int tid  = threadIdx.x;
  const int lane = tid & 31;
  const int w    = tid >> 5;
  const int hf   = lane >> 4;
  const int l15  = lane & 15;
  const int m0   = blockIdx.x * 128;
  const int n0   = blockIdx.y * 64;
  const int mw   = (w & 3) * 32;    // wave m-offset in block tile
  const int nw   = (w >> 2) * 32;   // wave n-offset in block tile

  const ushort_t* Ab = xb  + (size_t)m0 * E_;
  const ushort_t* Bb = WaT + (size_t)n0 * E_;

  v8f acc[2][2] = {{{}, {}}, {{}, {}}};

  if (w == 0) {  // TDM ignores EXEC: gate by wave, one issue per workgroup
    tdm_load_2d(lds_off(&At[0][0]), Ab, E_, M_, 32, 128, E_);
    tdm_load_2d(lds_off(&Bt[0][0]), Bb, E_, E3_, 32, 64, E_);
  }

  const int nk = E_ / 32;
  for (int kt = 0; kt < nk; ++kt) {
    const int cur = kt & 1;
    if (w == 0) {
      if (kt + 1 < nk) {   // prefetch next tile into other buffer
        tdm_load_2d(lds_off(&At[cur ^ 1][0]), Ab + (kt + 1) * 32, E_, M_, 32, 128, E_);
        tdm_load_2d(lds_off(&Bt[cur ^ 1][0]), Bb + (kt + 1) * 32, E_, E3_, 32, 64, E_);
        __builtin_amdgcn_s_wait_tensorcnt(2);  // current tile done, next in flight
      } else {
        __builtin_amdgcn_s_wait_tensorcnt(0);
      }
    }
    __syncthreads();

    const u32x4* A = At[cur];
    const u32x4* B = Bt[cur];
    v16bf b0 = mkfrag(B[(nw + l15) * 4 + 2 * hf],      B[(nw + l15) * 4 + 2 * hf + 1]);
    v16bf b1 = mkfrag(B[(nw + 16 + l15) * 4 + 2 * hf], B[(nw + 16 + l15) * 4 + 2 * hf + 1]);
    v16bf a0 = mkfrag(A[(mw + l15) * 4 + 2 * hf],      A[(mw + l15) * 4 + 2 * hf + 1]);
    v16bf a1 = mkfrag(A[(mw + 16 + l15) * 4 + 2 * hf], A[(mw + 16 + l15) * 4 + 2 * hf + 1]);
    acc[0][0] = __builtin_amdgcn_wmma_f32_16x16x32_bf16(false, a0, false, b0, (short)0, acc[0][0], false, false);
    acc[0][1] = __builtin_amdgcn_wmma_f32_16x16x32_bf16(false, a0, false, b1, (short)0, acc[0][1], false, false);
    acc[1][0] = __builtin_amdgcn_wmma_f32_16x16x32_bf16(false, a1, false, b0, (short)0, acc[1][0], false, false);
    acc[1][1] = __builtin_amdgcn_wmma_f32_16x16x32_bf16(false, a1, false, b1, (short)0, acc[1][1], false, false);
    __syncthreads();  // tile consumed before TDM overwrites it
  }

#pragma unroll
  for (int c = 0; c < 2; ++c) {
    const int ncol = n0 + nw + c * 16 + l15;        // [0, 2304)
    const int a    = ncol / E_;                     // 0=q 1=k 2=v
    const int rem  = ncol - a * E_;
    const int h    = rem >> 6;
    const int hs   = rem & 63;
    const float bias = b_attn[ncol];
#pragma unroll
    for (int r = 0; r < 2; ++r) {
#pragma unroll
      for (int j = 0; j < 8; ++j) {
        int mrow = m0 + mw + r * 16 + j + 8 * hf;   // [0, 8192)
        int bb = mrow >> 12;
        int t  = mrow & (T_ - 1);
        float val = acc[r][c][j] + bias;
        size_t idx;
        if (a == 2)  // V transposed: [B][NH][HS][T]
          idx = 2 * QKV_SLOT + (((size_t)(bb * NH_ + h) * HS_ + hs) * T_ + t);
        else         // Q/K: [B][NH][T][HS]
          idx = (size_t)a * QKV_SLOT + (((size_t)(bb * NH_ + h) * T_ + t) * HS_ + hs);
        qkv[idx] = f2bf(val);
      }
    }
  }
}

// ---------------- flash attention ----------------
// One wave per (b, h, 16-query tile). Key tiles of 32.
__global__ __launch_bounds__(256) void k_attn(const ushort_t* __restrict__ qkv,
                                              ushort_t* __restrict__ y) {
  __shared__ __align__(16) ushort_t P[8 * 16 * 32];  // per-wave P tile
  const int tid  = threadIdx.x;
  const int lane = tid & 31;
  const int w    = tid >> 5;
  const int hf   = lane >> 4;
  const int l15  = lane & 15;
  ushort_t* Pw   = P + w * (16 * 32);

  const int gw = blockIdx.x * 8 + w;
  const int qt = gw & 255;                 // T_/16 = 256 q-tiles
  const int h  = (gw >> 8) % NH_;
  const int b  = gw / (256 * NH_);
  const int q0 = qt * 16;

  const ushort_t* qbase = qkv + (((size_t)(b * NH_ + h) * T_) + q0) * HS_;
  const ushort_t* kbase = qkv + QKV_SLOT + ((size_t)(b * NH_ + h) * T_) * HS_;
  const ushort_t* vT    = qkv + 2 * QKV_SLOT + ((size_t)(b * NH_ + h) * HS_) * T_;

  const ushort_t* qrow = qbase + (size_t)l15 * HS_;
  v16bf qf0 = mkfrag(*(const u32x4*)(qrow + 16 * hf),
                     *(const u32x4*)(qrow + 16 * hf + 8));
  v16bf qf1 = mkfrag(*(const u32x4*)(qrow + 32 + 16 * hf),
                     *(const u32x4*)(qrow + 32 + 16 * hf + 8));

  v8f o[4] = {{}, {}, {}, {}};
  float mrow[8], lrow[8];
#pragma unroll
  for (int j = 0; j < 8; ++j) { mrow[j] = -__builtin_inff(); lrow[j] = 0.f; }

  const int nk = (q0 + 16 + 31) >> 5;
  for (int kt = 0; kt < nk; ++kt) {
    const int k0 = kt * 32;
    v8f s0 = {}, s1 = {};
    {
      const ushort_t* kr0 = kbase + (size_t)(k0 + l15) * HS_;
      v16bf kf00 = mkfrag(*(const u32x4*)(kr0 + 16 * hf),
                          *(const u32x4*)(kr0 + 16 * hf + 8));
      v16bf kf01 = mkfrag(*(const u32x4*)(kr0 + 32 + 16 * hf),
                          *(const u32x4*)(kr0 + 32 + 16 * hf + 8));
      s0 = __builtin_amdgcn_wmma_f32_16x16x32_bf16(false, qf0, false, kf00,
                                                   (short)0, s0, false, false);
      s0 = __builtin_amdgcn_wmma_f32_16x16x32_bf16(false, qf1, false, kf01,
                                                   (short)0, s0, false, false);
      const ushort_t* kr1 = kbase + (size_t)(k0 + 16 + l15) * HS_;
      v16bf kf10 = mkfrag(*(const u32x4*)(kr1 + 16 * hf),
                          *(const u32x4*)(kr1 + 16 * hf + 8));
      v16bf kf11 = mkfrag(*(const u32x4*)(kr1 + 32 + 16 * hf),
                          *(const u32x4*)(kr1 + 32 + 16 * hf + 8));
      s1 = __builtin_amdgcn_wmma_f32_16x16x32_bf16(false, qf0, false, kf10,
                                                   (short)0, s1, false, false);
      s1 = __builtin_amdgcn_wmma_f32_16x16x32_bf16(false, qf1, false, kf11,
                                                   (short)0, s1, false, false);
    }

#pragma unroll
    for (int j = 0; j < 8; ++j) {
      const int qg = q0 + j + 8 * hf;
      float v0 = s0[j] * 0.125f;       // 1/sqrt(64)
      float v1 = s1[j] * 0.125f;
      if (k0 + l15 > qg)      v0 = -__builtin_inff();
      if (k0 + 16 + l15 > qg) v1 = -__builtin_inff();
      float mx = fmaxf(v0, v1);
#pragma unroll
      for (int off = 1; off < 16; off <<= 1)
        mx = fmaxf(mx, __shfl_xor(mx, off, 32));
      float mnew  = fmaxf(mrow[j], mx);
      float scale = __expf(mrow[j] - mnew);
      float p0 = __expf(v0 - mnew);
      float p1 = __expf(v1 - mnew);
      float rs = p0 + p1;
#pragma unroll
      for (int off = 1; off < 16; off <<= 1)
        rs += __shfl_xor(rs, off, 32);
      lrow[j] = lrow[j] * scale + rs;
      mrow[j] = mnew;
      o[0][j] *= scale; o[1][j] *= scale; o[2][j] *= scale; o[3][j] *= scale;
      const int pr = j + 8 * hf;
      Pw[pr * 32 + l15]      = f2bf(p0);
      Pw[pr * 32 + 16 + l15] = f2bf(p1);
    }
    asm volatile("s_wait_dscnt 0" ::: "memory");  // same-wave LDS ordering

    v16bf pf = mkfrag(*(const u32x4*)(Pw + l15 * 32 + 16 * hf),
                      *(const u32x4*)(Pw + l15 * 32 + 16 * hf + 8));
#pragma unroll
    for (int nt = 0; nt < 4; ++nt) {
      const ushort_t* vr = vT + (size_t)(nt * 16 + l15) * T_ + k0;
      v16bf vf = mkfrag(*(const u32x4*)(vr + 16 * hf),
                        *(const u32x4*)(vr + 16 * hf + 8));
      o[nt] = __builtin_amdgcn_wmma_f32_16x16x32_bf16(false, pf, false, vf,
                                                      (short)0, o[nt], false, false);
    }
    asm volatile("s_wait_dscnt 0" ::: "memory");
  }

#pragma unroll
  for (int j = 0; j < 8; ++j) {
    float inv = lrow[j] > 0.f ? 1.0f / lrow[j] : 0.f;
    const int trow = q0 + j + 8 * hf;
    size_t base = ((size_t)b * T_ + trow) * E_ + h * HS_ + l15;
    y[base + 0]  = f2bf(o[0][j] * inv);
    y[base + 16] = f2bf(o[1][j] * inv);
    y[base + 32] = f2bf(o[2][j] * inv);
    y[base + 48] = f2bf(o[3][j] * inv);
  }
}

// ---------------- output projection GEMM ----------------
__global__ __launch_bounds__(256) void k_proj_gemm(
    const ushort_t* __restrict__ yb, const ushort_t* __restrict__ WpT,
    const float* __restrict__ b_proj, float* __restrict__ out) {
  __shared__ __align__(16) u32x4 At[2][128 * 4];
  __shared__ __align__(16) u32x4 Bt[2][64 * 4];

  const int tid  = threadIdx.x;
  const int lane = tid & 31;
  const int w    = tid >> 5;
  const int hf   = lane >> 4;
  const int l15  = lane & 15;
  const int m0   = blockIdx.x * 128;
  const int n0   = blockIdx.y * 64;
  const int mw   = (w & 3) * 32;
  const int nw   = (w >> 2) * 32;

  const ushort_t* Ab = yb  + (size_t)m0 * E_;
  const ushort_t* Bb = WpT + (size_t)n0 * E_;

  v8f acc[2][2] = {{{}, {}}, {{}, {}}};

  if (w == 0) {
    tdm_load_2d(lds_off(&At[0][0]), Ab, E_, M_, 32, 128, E_);
    tdm_load_2d(lds_off(&Bt[0][0]), Bb, E_, E_, 32, 64, E_);
  }

  const int nk = E_ / 32;
  for (int kt = 0; kt < nk; ++kt) {
    const int cur = kt & 1;
    if (w == 0) {
      if (kt + 1 < nk) {
        tdm_load_2d(lds_off(&At[cur ^ 1][0]), Ab + (kt + 1) * 32, E_, M_, 32, 128, E_);
        tdm_load_2d(lds_off(&Bt[cur ^ 1][0]), Bb + (kt + 1) * 32, E_, E_, 32, 64, E_);
        __builtin_amdgcn_s_wait_tensorcnt(2);
      } else {
        __builtin_amdgcn_s_wait_tensorcnt(0);
      }
    }
    __syncthreads();

    const u32x4* A = At[cur];
    const u32x4* B = Bt[cur];
    v16bf b0 = mkfrag(B[(nw + l15) * 4 + 2 * hf],      B[(nw + l15) * 4 + 2 * hf + 1]);
    v16bf b1 = mkfrag(B[(nw + 16 + l15) * 4 + 2 * hf], B[(nw + 16 + l15) * 4 + 2 * hf + 1]);
    v16bf a0 = mkfrag(A[(mw + l15) * 4 + 2 * hf],      A[(mw + l15) * 4 + 2 * hf + 1]);
    v16bf a1 = mkfrag(A[(mw + 16 + l15) * 4 + 2 * hf], A[(mw + 16 + l15) * 4 + 2 * hf + 1]);
    acc[0][0] = __builtin_amdgcn_wmma_f32_16x16x32_bf16(false, a0, false, b0, (short)0, acc[0][0], false, false);
    acc[0][1] = __builtin_amdgcn_wmma_f32_16x16x32_bf16(false, a0, false, b1, (short)0, acc[0][1], false, false);
    acc[1][0] = __builtin_amdgcn_wmma_f32_16x16x32_bf16(false, a1, false, b0, (short)0, acc[1][0], false, false);
    acc[1][1] = __builtin_amdgcn_wmma_f32_16x16x32_bf16(false, a1, false, b1, (short)0, acc[1][1], false, false);
    __syncthreads();
  }

#pragma unroll
  for (int c = 0; c < 2; ++c) {
    const int ncol = n0 + nw + c * 16 + l15;
    const float bias = b_proj[ncol];
#pragma unroll
    for (int r = 0; r < 2; ++r) {
#pragma unroll
      for (int j = 0; j < 8; ++j) {
        int mrow = m0 + mw + r * 16 + j + 8 * hf;
        out[(size_t)mrow * E_ + ncol] = acc[r][c][j] + bias;
      }
    }
  }
}

// ---------------- launcher ----------------
extern "C" void kernel_launch(void* const* d_in, const int* in_sizes, int n_in,
                              void* d_out, int out_size, void* d_ws, size_t ws_size,
                              hipStream_t stream) {
  (void)in_sizes; (void)n_in; (void)out_size; (void)ws_size;
  const float* x      = (const float*)d_in[0];
  const float* W_attn = (const float*)d_in[1];
  const float* b_attn = (const float*)d_in[2];
  const float* W_proj = (const float*)d_in[3];
  const float* b_proj = (const float*)d_in[4];
  float* out = (float*)d_out;

  char* ws = (char*)d_ws;
  const size_t o_xb  = 0;                                   // M_*E_ bf16
  const size_t o_WaT = o_xb  + (size_t)M_ * E_ * 2;         // E3_*E_ bf16
  const size_t o_WpT = o_WaT + (size_t)E3_ * E_ * 2;        // E_*E_ bf16
  const size_t o_qkv = o_WpT + (size_t)E_ * E_ * 2;         // 3 slots bf16
  const size_t o_y   = o_qkv + 3 * QKV_SLOT * 2;            // M_*E_ bf16

  ushort_t* xb   = (ushort_t*)(ws + o_xb);
  ushort_t* WaT  = (ushort_t*)(ws + o_WaT);
  ushort_t* WpT  = (ushort_t*)(ws + o_WpT);
  ushort_t* qkvb = (ushort_t*)(ws + o_qkv);
  ushort_t* yb   = (ushort_t*)(ws + o_y);

  k_f32_to_bf16<<<4096, 256, 0, stream>>>(x, xb, M_ * E_);
  k_transpose_bf16<<<6912, 256, 0, stream>>>(W_attn, WaT, E_, E3_);
  k_transpose_bf16<<<2304, 256, 0, stream>>>(W_proj, WpT, E_, E_);

  k_qkv_gemm<<<dim3(M_ / 128, E3_ / 64), 256, 0, stream>>>(xb, WaT, b_attn, qkvb);
  k_attn<<<(B_ * NH_ * (T_ / 16)) / 8, 256, 0, stream>>>(qkvb, yb);
  k_proj_gemm<<<dim3(M_ / 128, E_ / 64), 256, 0, stream>>>(yb, WpT, b_proj, out);
}
